// SparseDeformableMambaBlock_39633958208119
// MI455X (gfx1250) — matrix operand order repaired and checked by reading, not compiled, not run on hardware
//
#include <hip/hip_runtime.h>
#include <hip/hip_bf16.h>
#include <math.h>

#define BATCH  8
#define SEQ    2048
#define DIM    512
#define EDIM   1024
#define KSEL   614                 // int(2048 * 0.3)
#define M1     (BATCH * SEQ)       // 16384 rows for GEMM1
#define KT1    (DIM / 32)          // 16 k-tiles (K=512)
#define NT1    (EDIM / 16)         // 64 n-tiles
#define M2     (BATCH * KSEL)      // 4912 rows for GEMM2 (= 307 * 16, exact)
#define MT2    (M2 / 16)           // 307
#define KT2    (EDIM / 32)         // 32 k-tiles (K=1024)
#define NT2    (DIM / 16)          // 32 n-tiles

typedef __attribute__((ext_vector_type(16))) __bf16 v16bf;
typedef __attribute__((ext_vector_type(8)))  float  v8f;

union FragU { uint4 u[2]; v16bf bf; };

// round-to-nearest-even float -> bf16 bits
__device__ __forceinline__ unsigned short f2bf(float f) {
    unsigned int u = __float_as_uint(f);
    unsigned int r = (u + 0x7FFFu + ((u >> 16) & 1u)) >> 16;
    return (unsigned short)r;
}

// ---- WMMA fragment addressing (CDNA5 ISA 7.12.2, wave32) ----
// A (16x32 bf16): lane = (m&15) | (16 if (k%32)&15 >= 8); elem = (k&7) | (8 if k%32>=16)
__device__ __forceinline__ size_t a_frag_addr(int m, int k, int KT) {
    int kt = k >> 5, kk = k & 31;
    int lane = (m & 15) | (((kk & 15) >= 8) ? 16 : 0);
    int elem = (kk & 7) | ((kk >= 16) ? 8 : 0);
    return (((size_t)(m >> 4) * KT + kt) * 32 + lane) * 16 + elem;
}
// B (32x16 bf16): lane = (n&15) | (16 if (k%32)&15 >= 8); elem = (k&7) | (8 if k%32>=16)
__device__ __forceinline__ size_t b_frag_addr(int k, int n, int NT) {
    int kt = k >> 5, kk = k & 31;
    int lane = (n & 15) | (((kk & 15) >= 8) ? 16 : 0);
    int elem = (kk & 7) | ((kk >= 16) ? 8 : 0);
    return (((size_t)kt * NT + (n >> 4)) * 32 + lane) * 16 + elem;
}

// ---- pipelined 1 m-tile x 4 n-tile WMMA inner loop (double-buffered fragments) ----
template<int KT, int NT>
__device__ __forceinline__ void wmma_tile_1x4(const unsigned short* __restrict__ Ap,
                                              const unsigned short* __restrict__ Bp,
                                              int mtile, int ng, int lane, v8f acc[4]) {
    FragU a0, a1, b0[4], b1[4];
    auto ldA = [&](int kt, FragU& f) {
        const uint4* p = (const uint4*)(Ap + (((size_t)mtile * KT + kt) * 32 + lane) * 16);
        f.u[0] = p[0]; f.u[1] = p[1];
    };
    auto ldB = [&](int kt, int j, FragU& f) {
        const uint4* p = (const uint4*)(Bp + (((size_t)kt * NT + (ng * 4 + j)) * 32 + lane) * 16);
        f.u[0] = p[0]; f.u[1] = p[1];
    };
    ldA(0, a0);
#pragma unroll
    for (int j = 0; j < 4; ++j) ldB(0, j, b0[j]);
#pragma unroll 2
    for (int kt = 0; kt < KT; ++kt) {
        FragU& ac = (kt & 1) ? a1 : a0;
        FragU* bc = (kt & 1) ? b1 : b0;
        FragU& an = (kt & 1) ? a0 : a1;
        FragU* bn = (kt & 1) ? b0 : b1;
        if (kt + 1 < KT) {
            ldA(kt + 1, an);                       // issue next tile's loads first,
            __builtin_prefetch(Ap + (((size_t)mtile * KT + kt + 2) * 32 + lane) * 16, 0, 3);
#pragma unroll
            for (int j = 0; j < 4; ++j) ldB(kt + 1, j, bn[j]);
        }
#pragma unroll
        for (int j = 0; j < 4; ++j)                // ...then consume current tile
            acc[j] = __builtin_amdgcn_wmma_f32_16x16x32_bf16(
                false, ac.bf, false, bc[j].bf, (short)0, acc[j], false, false);
    }
}

// ---------------- K0: pack weight matrix (K x N, row major f32) to B fragments ----------------
__global__ void pack_weights_kernel(const float* __restrict__ W, unsigned short* __restrict__ Bp,
                                    int K, int N) {
    int i = blockIdx.x * 256 + threadIdx.x;
    if (i >= K * N) return;
    int k = i / N, n = i - k * N;
    Bp[b_frag_addr(k, n, N >> 4)] = f2bf(W[i]);
}

// ---------------- K1: RMS-norm a row of x and pack bf16 into A1 fragments ----------------
__global__ void rmsnorm_pack_kernel(const float* __restrict__ x, const float* __restrict__ nw,
                                    unsigned short* __restrict__ Ap) {
    int m = blockIdx.x;                 // row (b*SEQ + l), 16384 blocks x 128 threads
    int tid = threadIdx.x;
    const float* row = x + (size_t)m * DIM;
    float v[4]; float s = 0.f;
#pragma unroll
    for (int i = 0; i < 4; ++i) { v[i] = row[tid + i * 128]; s += v[i] * v[i]; }
#pragma unroll
    for (int o = 16; o > 0; o >>= 1) s += __shfl_xor(s, o, 32);
    __shared__ float wsum[4];
    if ((tid & 31) == 0) wsum[tid >> 5] = s;
    __syncthreads();
    float tot = wsum[0] + wsum[1] + wsum[2] + wsum[3];
    float rms = sqrtf(tot) * 0.044194173824159216f;      // * 512^-0.5
    float inv = 1.0f / (rms + 1e-6f);
#pragma unroll
    for (int i = 0; i < 4; ++i) {
        int k = tid + i * 128;
        Ap[a_frag_addr(m, k, KT1)] = f2bf(v[i] * nw[k] * inv);
    }
}

// ---------------- K2: GEMM1  x_proj = x_norm @ W_in + b_in  (WMMA bf16) ----------------
__global__ void gemm1_kernel(const unsigned short* __restrict__ Ap,
                             const unsigned short* __restrict__ Bp,
                             const float* __restrict__ bias, float* __restrict__ out) {
    int wave = threadIdx.x >> 5, lane = threadIdx.x & 31;
    int job = blockIdx.x * 4 + wave;          // 16384 jobs
    int mtile = job >> 4;
    int ng = job & 15;                        // n-tile group: tiles ng*4 .. ng*4+3
    v8f zero = {0.f,0.f,0.f,0.f,0.f,0.f,0.f,0.f};
    v8f acc[4] = {zero, zero, zero, zero};
    wmma_tile_1x4<KT1, NT1>(Ap, Bp, mtile, ng, lane, acc);
    int n = lane & 15, mo = (lane & 16) ? 8 : 0;
#pragma unroll
    for (int j = 0; j < 4; ++j) {
        int col = (ng * 4 + j) * 16 + n;
        float bs = bias[col];
#pragma unroll
        for (int r = 0; r < 8; ++r)
            out[(size_t)(mtile * 16 + r + mo) * EDIM + col] = acc[j][r] + bs;
    }
}

// ---------------- K3: similarity logits (ranking-equivalent to softmax(cos)) ----------------
__global__ void sim_kernel(const float* __restrict__ xproj, float* __restrict__ sim) {
    int m = blockIdx.x;                 // 16384 blocks x 256 threads
    int b = m >> 11;
    const float* row = xproj + (size_t)m * EDIM;
    const float* cen = xproj + ((size_t)(b * SEQ + SEQ / 2)) * EDIM;
    float d = 0.f, n2 = 0.f;
    for (int i = threadIdx.x; i < EDIM; i += 256) {
        float r = row[i];
        d += r * cen[i];
        n2 += r * r;
    }
#pragma unroll
    for (int o = 16; o > 0; o >>= 1) { d += __shfl_xor(d, o, 32); n2 += __shfl_xor(n2, o, 32); }
    __shared__ float sd[8], sn[8];
    if ((threadIdx.x & 31) == 0) { sd[threadIdx.x >> 5] = d; sn[threadIdx.x >> 5] = n2; }
    __syncthreads();
    if (threadIdx.x == 0) {
        float D = 0.f, N = 0.f;
#pragma unroll
        for (int w = 0; w < 8; ++w) { D += sd[w]; N += sn[w]; }
        sim[m] = D / fmaxf(sqrtf(N), 1e-12f);
    }
}

// ---------------- K4: per-batch exact top-k via full bitonic sort (desc value, asc index) ----------------
__global__ void topk_kernel(const float* __restrict__ sim, int* __restrict__ idx) {
    __shared__ float v[SEQ];
    __shared__ int   id[SEQ];
    int b = blockIdx.x, t = threadIdx.x;          // 1024 threads
    for (int i = t; i < SEQ; i += 1024) { v[i] = sim[b * SEQ + i]; id[i] = i; }
    __syncthreads();
    for (int size = 2; size <= SEQ; size <<= 1) {
        for (int stride = size >> 1; stride > 0; stride >>= 1) {
            int j = ((t & ~(stride - 1)) << 1) | (t & (stride - 1));
            int pj = j | stride;
            bool desc = ((j & size) == 0);
            float va = v[j], vb = v[pj];
            int ia = id[j], ib = id[pj];
            bool b_gt_a = (vb > va) || ((vb == va) && (ib < ia));
            bool doswap = desc ? b_gt_a : ((va > vb) || ((va == vb) && (ia < ib)));
            if (doswap) { v[j] = vb; v[pj] = va; id[j] = ib; id[pj] = ia; }
            __syncthreads();
        }
    }
    for (int i = t; i < KSEL; i += 1024) idx[b * KSEL + i] = id[i];
}

// ---------------- K5: gather sparse rows + depthwise causal conv (taps=4) ----------------
__global__ void conv_gather_kernel(const float* __restrict__ xproj, const int* __restrict__ topk,
                                   const float* __restrict__ cw, float* __restrict__ xconv) {
    int bt = blockIdx.x;                 // 4912 blocks x 256 threads
    int b = bt / KSEL, t = bt - b * KSEL;
    int rows[4];
#pragma unroll
    for (int j = 0; j < 4; ++j) {
        int tt = t - 3 + j;
        rows[j] = (tt >= 0) ? topk[b * KSEL + tt] : -1;
    }
    for (int e = threadIdx.x; e < EDIM; e += 256) {
        float acc = 0.f;
#pragma unroll
        for (int j = 0; j < 4; ++j)
            if (rows[j] >= 0)
                acc += cw[e * 4 + j] * xproj[((size_t)(b * SEQ + rows[j])) * EDIM + e];
        xconv[(size_t)bt * EDIM + e] = acc;
    }
}

// ---------------- K6: SSM scan; writes y directly as GEMM2 A-fragments (bf16) ----------------
__global__ void scan_kernel(const float* __restrict__ xconv, const float* __restrict__ A,
                            const float* __restrict__ Bpar, const float* __restrict__ Cpar,
                            unsigned short* __restrict__ Ap2) {
    __shared__ float As[256];
    __shared__ float sB[16];
    int tid = threadIdx.x;
    As[tid] = A[tid & 255];                      // 256 threads cover the 16x16 A
    if (tid < 16) sB[tid] = 1.0f / (1.0f + expf(-Bpar[tid]));
    __syncthreads();
    int g = blockIdx.x * 256 + tid;              // 8192 channels total
    int b = g >> 10, e = g & 1023;
    float sC[16];
#pragma unroll
    for (int s = 0; s < 16; ++s) sC[s] = 1.0f / (1.0f + expf(-Cpar[e * 16 + s]));
    float h[16];
#pragma unroll
    for (int s = 0; s < 16; ++s) h[s] = 0.f;
    // fragment coordinates for column k=e (fixed across the time loop)
    int kt = e >> 5, kk = e & 31;
    int lane_off = (((kk & 15) >= 8) ? 16 : 0);
    int elem = (kk & 7) | ((kk >= 16) ? 8 : 0);
    for (int t = 0; t < KSEL; ++t) {
        float xv = xconv[((size_t)(b * KSEL + t)) * EDIM + e];
        float hn[16];
#pragma unroll
        for (int r = 0; r < 16; ++r) {
            float a = sB[r] * xv;
#pragma unroll
            for (int s = 0; s < 16; ++s) a += As[r * 16 + s] * h[s];
            hn[r] = a;
        }
        float y = 0.f;
#pragma unroll
        for (int r = 0; r < 16; ++r) { h[r] = hn[r]; y += hn[r] * sC[r]; }
        int m = b * KSEL + t;
        size_t addr = (((size_t)(m >> 4) * KT2 + kt) * 32 + ((m & 15) | lane_off)) * 16 + elem;
        Ap2[addr] = f2bf(y);
    }
}

// ---------------- K7: GEMM2 + bias + residual scatter into d_out (WMMA bf16) ----------------
__global__ void gemm2_kernel(const unsigned short* __restrict__ Ap,
                             const unsigned short* __restrict__ Bp,
                             const float* __restrict__ bias, const int* __restrict__ topk,
                             const float* __restrict__ x, float* __restrict__ out) {
    int wave = threadIdx.x >> 5, lane = threadIdx.x & 31;
    int job = blockIdx.x * 4 + wave;          // 2456 jobs = 307 mtiles * 8 ngroups
    int mtile = job >> 3;
    int ng = job & 7;
    v8f zero = {0.f,0.f,0.f,0.f,0.f,0.f,0.f,0.f};
    v8f acc[4] = {zero, zero, zero, zero};
    wmma_tile_1x4<KT2, NT2>(Ap, Bp, mtile, ng, lane, acc);
    int n = lane & 15, mo = (lane & 16) ? 8 : 0;
#pragma unroll
    for (int j = 0; j < 4; ++j) {
        int col = (ng * 4 + j) * 16 + n;
        float bs = bias[col];
#pragma unroll
        for (int r = 0; r < 8; ++r) {
            int m = mtile * 16 + r + mo;       // 0..4911, all valid
            int b = m / KSEL, t = m - b * KSEL;
            int l = topk[b * KSEL + t];
            size_t pos = ((size_t)(b * SEQ + l)) * DIM + col;
            out[pos] = x[pos] + acc[j][r] + bs;    // full overwrite (replay-safe)
        }
    }
}

// ---------------- host side ----------------
extern "C" void kernel_launch(void* const* d_in, const int* in_sizes, int n_in,
                              void* d_out, int out_size, void* d_ws, size_t ws_size,
                              hipStream_t stream) {
    const float* x      = (const float*)d_in[0];
    const float* norm_w = (const float*)d_in[1];
    const float* W_in   = (const float*)d_in[2];
    const float* b_in   = (const float*)d_in[3];
    const float* W_out  = (const float*)d_in[4];
    const float* b_out  = (const float*)d_in[5];
    const float* Amat   = (const float*)d_in[6];
    const float* Bp     = (const float*)d_in[7];
    const float* Cp     = (const float*)d_in[8];
    const float* conv_w = (const float*)d_in[9];
    float* out = (float*)d_out;

    // workspace layout (256B aligned regions)
    char* w = (char*)d_ws;
    float*          xproj  = (float*)(w);                                  // 16384*1024*4 = 64 MiB
    unsigned short* apack1 = (unsigned short*)(w + 67108864);              // 16384*512*2  = 16 MiB
    unsigned short* bpack1 = (unsigned short*)(w + 67108864 + 16777216);   // 512*1024*2   = 1 MiB
    unsigned short* bpack2 = (unsigned short*)(w + 67108864 + 16777216 + 1048576);
    float*          simv   = (float*)(w + 67108864 + 16777216 + 2097152);  // 16384*4
    int*            topk   = (int*)(w + 67108864 + 16777216 + 2097152 + 65536);   // 8*614*4
    float*          xconv  = (float*)(w + 67108864 + 16777216 + 2097152 + 65536 + 32768); // 4912*1024*4
    unsigned short* apack2 = (unsigned short*)(w + 67108864 + 16777216 + 2097152 + 65536 + 32768 + 20119552);

    // residual: out starts as x everywhere (scatter positions overwritten by gemm2)
    hipMemcpyAsync(out, x, (size_t)M1 * DIM * sizeof(float), hipMemcpyDeviceToDevice, stream);

    pack_weights_kernel<<<(DIM * EDIM + 255) / 256, 256, 0, stream>>>(W_in, bpack1, DIM, EDIM);
    pack_weights_kernel<<<(EDIM * DIM + 255) / 256, 256, 0, stream>>>(W_out, bpack2, EDIM, DIM);

    rmsnorm_pack_kernel<<<M1, 128, 0, stream>>>(x, norm_w, apack1);

    gemm1_kernel<<<(M1 / 16) * (NT1 / 4) / 4, 128, 0, stream>>>(apack1, bpack1, b_in, xproj);

    sim_kernel<<<M1, 256, 0, stream>>>(xproj, simv);

    topk_kernel<<<BATCH, 1024, 0, stream>>>(simv, topk);

    conv_gather_kernel<<<M2, 256, 0, stream>>>(xproj, topk, conv_w, xconv);

    scan_kernel<<<(BATCH * EDIM) / 256, 256, 0, stream>>>(xconv, Amat, Bp, Cp, apack2);

    gemm2_kernel<<<(MT2 * (NT2 / 4)) / 4, 128, 0, stream>>>(apack2, bpack2, b_out, topk, x, out);
}
